// CGP_8899172237465
// MI455X (gfx1250) — matrix operand order repaired and checked by reading, not compiled, not run on hardware
//
#include <hip/hip_runtime.h>
#include <hip/hip_bf16.h>

// ---------------------------------------------------------------------------
// Fused GCN-ODE as a single operator:  out_n = T (1984x1984) @ x_n (1984x512)
//   T[(o,v),(c,w)] = sum_j mlp_w[o, j*32+c] * (M^j)[v,w],  M = 0.75 I + 0.25 A_norm
// Main GEMM: v_wmma_f32_16x16x32_bf16 fed from LDS, filled with
// global_load_async_to_lds_b128 (ASYNCcnt) in a double-buffered pipeline.
// Operands are pre-swizzled into exact WMMA fragment lane layouts and padded
// (K: 62->64 fragment steps, M: 124->128 tiles) with zeros.
// ---------------------------------------------------------------------------

typedef __attribute__((ext_vector_type(16))) __bf16 v16bf;
typedef __attribute__((ext_vector_type(8)))  float  v8f;
typedef __attribute__((ext_vector_type(4)))  int    v4i;

#define AS1 __attribute__((address_space(1)))
#define AS3 __attribute__((address_space(3)))

static constexpr int NV    = 62;          // nodes
static constexpr int NV2   = NV * NV;
static constexpr int NHID  = 30;          // gate hidden
static constexpr int NC    = 32;          // channels
static constexpr int NL    = 512;         // sequence length
static constexpr int NMROW = NC * NV;     // 1984 = M = K of the big GEMM
static constexpr int NMT   = NMROW / 16;  // 124 valid M-tiles
static constexpr int NMTP  = 128;         // padded M-tiles
static constexpr int NKS   = NMROW / 32;  // 62 valid k-steps
static constexpr int NKSP  = 64;          // padded k-steps
static constexpr int NNT   = NL / 16;     // 32 N-tiles per batch

#if defined(__AMDGCN__) && __has_builtin(__builtin_amdgcn_global_load_async_to_lds_b128)
#define HAVE_ASYNC 1
#else
#define HAVE_ASYNC 0
#endif

static __device__ inline unsigned short f2bf(float f) {
  unsigned int u = __float_as_uint(f);
  u += 0x7FFFu + ((u >> 16) & 1u);           // round-to-nearest-even
  return (unsigned short)(u >> 16);
}

// --- gate MLP: adj_buf(1x3844) -> ELU(30) -> tanh/relu(3844) ----------------
__global__ void gate_kernel(const float* __restrict__ adj_buf,
                            const float* __restrict__ w1,
                            const float* __restrict__ w2,
                            float* __restrict__ adj_ds) {
  __shared__ float hid[32];
  const int tid = threadIdx.x, lane = tid & 31, wave = tid >> 5;
  for (int h = wave; h < NHID; h += 8) {
    float s = 0.f;
    for (int k = lane; k < NV2; k += 32) s += adj_buf[k] * w1[h * NV2 + k];
    for (int off = 16; off > 0; off >>= 1) s += __shfl_down(s, off, 32);
    if (lane == 0) hid[h] = (s > 0.f) ? s : (expf(s) - 1.f);   // ELU
  }
  __syncthreads();
  for (int k = tid; k < NV2; k += 256) {
    float z = 0.f;
#pragma unroll
    for (int h = 0; h < NHID; ++h) z += hid[h] * w2[k * NHID + h];
    float t = tanhf(z);
    adj_ds[k] = t > 0.f ? t : 0.f;                             // relu(tanh)
  }
}

// --- adj = PLI @ adj_ds + I; degree-normalize; M = 0.75 I + 0.25 A_norm -----
__global__ void adjm_kernel(const float* __restrict__ pli,
                            const float* __restrict__ ds,
                            float* __restrict__ Mout) {
  __shared__ float sA[NV2];
  __shared__ float dinv[NV];
  const int tid = threadIdx.x;
  for (int e = tid; e < NV2; e += 256) {
    int i = e / NV, j = e - i * NV;
    float s = (i == j) ? 1.0f : 0.0f;
    for (int k = 0; k < NV; ++k) s += pli[i * NV + k] * ds[k * NV + j];
    sA[e] = s;
  }
  __syncthreads();
  if (tid < NV) {
    float s = 0.f;
    for (int j = 0; j < NV; ++j) s += sA[tid * NV + j];
    dinv[tid] = rsqrtf(s);
  }
  __syncthreads();
  for (int e = tid; e < NV2; e += 256) {
    int i = e / NV, j = e - i * NV;
    Mout[e] = 0.75f * (i == j ? 1.f : 0.f) + 0.25f * dinv[i] * sA[e] * dinv[j];
  }
}

// --- M powers: Pg[j-1] = M^j for j = 1..4 -----------------------------------
__global__ void pow_kernel(const float* __restrict__ Mm, float* __restrict__ Pg) {
  __shared__ float sM[NV2];
  __shared__ float sP[NV2];
  const int tid = threadIdx.x;
  for (int e = tid; e < NV2; e += 256) { float m = Mm[e]; sM[e] = m; sP[e] = m; Pg[e] = m; }
  __syncthreads();
  for (int step = 2; step <= 4; ++step) {
    float tmp[16];
    int c = 0;
    for (int e = tid; e < NV2; e += 256) {
      int i = e / NV, j = e - i * NV;
      float s = 0.f;
      for (int k = 0; k < NV; ++k) s += sM[i * NV + k] * sP[k * NV + j];
      tmp[c++] = s;
    }
    __syncthreads();
    c = 0;
    for (int e = tid; e < NV2; e += 256) { sP[e] = tmp[c]; Pg[(step - 1) * 4096 + e] = tmp[c]; ++c; }
    __syncthreads();
  }
}

// --- build T in bf16, pre-swizzled 16x32 A-fragment layout, zero-padded -----
// A frag (ISA 7.12.2): lane r=lane&15, hi=lane>>4; element e=2*vreg+pair holds
//   A[r, (vreg>>2)*16 + hi*8 + (vreg&3)*2 + pair]
// Storage: Tf[(mt*64 + ks)*512 + lane*16 + e], mt<128, ks<64 (pad -> 0)
__global__ void build_t(const float* __restrict__ Pg,
                        const float* __restrict__ mlp_w,
                        unsigned short* __restrict__ Tf) {
  size_t idx = (size_t)blockIdx.x * blockDim.x + threadIdx.x;
  if (idx >= (size_t)NMTP * NKSP * 512) return;
  int e    = (int)(idx & 15);
  int lane = (int)((idx >> 4) & 31);
  size_t t = idx >> 9;
  int ks = (int)(t & 63);
  int mt = (int)(t >> 6);
  if (ks >= NKS || mt >= NMT) { Tf[idx] = 0; return; }
  int r = lane & 15, hi = lane >> 4;
  int vreg = e >> 1, pair = e & 1;
  int kl = (vreg >> 2) * 16 + hi * 8 + (vreg & 3) * 2 + pair;
  int row = mt * 16 + r;          // = o*62 + v
  int col = ks * 32 + kl;         // = c*62 + w
  int o = row / NV, vn = row - o * NV;
  int c = col / NV, w  = col - c * NV;
  float acc = (vn == w) ? mlp_w[o * (5 * NC) + c] : 0.0f;   // j=0 term (M^0 = I)
#pragma unroll
  for (int j = 1; j <= 4; ++j)
    acc += mlp_w[o * (5 * NC) + j * NC + c] * Pg[(j - 1) * 4096 + vn * NV + w];
  Tf[idx] = f2bf(acc);
}

// --- repack x to bf16, 32x16 B-fragment layout, zero-padded -----------------
// B frag: lane n=lane&15, hi=lane>>4; element e holds B[(e>>3)*16+hi*8+(e&7), n]
// Storage: Bf[((n*32+nt)*64 + ks)*512 + lane*16 + e], ks<64 (pad -> 0)
__global__ void swizzle_b(const float* __restrict__ x, unsigned short* __restrict__ Bf) {
  size_t idx = (size_t)blockIdx.x * blockDim.x + threadIdx.x;   // 32*32*64*512
  int e    = (int)(idx & 15);
  int lane = (int)((idx >> 4) & 31);
  size_t t = idx >> 9;
  int ks = (int)(t & 63);
  size_t t2 = t >> 6;
  int nt = (int)(t2 & 31);
  int n  = (int)(t2 >> 5);
  if (ks >= NKS) { Bf[idx] = 0; return; }
  int hi = lane >> 4;
  int kg = ks * 32 + (e >> 3) * 16 + hi * 8 + (e & 7);   // global K row = c*62 + w
  int c = kg / NV;
  int w = kg - c * NV;
  int l = nt * 16 + (lane & 15);
  float val = x[(((size_t)n * NC + c) * NV + w) * NL + l];
  Bf[idx] = f2bf(val);
}

// --- main batched GEMM: async global->LDS double-buffered WMMA pipeline -----
// grid: (16, 8, 32); block = 8 waves. Block tile 128M x 64N; wave grid 4Mx2N,
// each wave 2x2 C tiles. Chunk = 4 k-steps (K=128): A 32KB + B 16KB per buffer.
__global__ __launch_bounds__(256) void gemm_wmma(
    const unsigned short* __restrict__ Tf,
    const unsigned short* __restrict__ Bf,
    const float* __restrict__ bias,
    float* __restrict__ out) {
  extern __shared__ char lds[];               // 2 * 49152 bytes
  const int tid  = threadIdx.x;
  const int lane = tid & 31;
  const int wave = tid >> 5;
  const int mw = wave & 3, nw = wave >> 2;
  const int mtBase = blockIdx.x * 8;
  const int ntBase = blockIdx.y * 4;
  const int nz = blockIdx.z;
  const int mt0 = mtBase + mw * 2;
  const int nt0 = ntBase + nw * 2;

  const char* Ag0 = (const char*)Tf + (size_t)mtBase * (NKSP * 1024);
  const char* Bg0 = (const char*)Bf + (size_t)(nz * NNT + ntBase) * (NKSP * 1024);

  v8f c00 = {}, c01 = {}, c10 = {}, c11 = {};

  // stage chunk ch into buffer buf: 8 A streams + 4 B streams, 4KB each,
  // 3072 x 16B transfers spread over 256 threads (12 per thread)
  auto stage = [&](int ch, int buf) {
    const char* Ag = Ag0 + ch * 4096;
    const char* Bg = Bg0 + ch * 4096;
    char* Lb = lds + buf * 49152;
    int c = tid;
#pragma unroll
    for (int it = 0; it < 12; ++it, c += 256) {
      int isB = c >> 11;                    // 0: A (c<2048), 1: B
      int cc  = c & 2047;
      int s   = cc >> 8;                    // stream within group
      int off = (cc & 255) << 4;            // byte offset in stream chunk
      const char* g = (isB ? Bg : Ag) + ((size_t)s << 16) + off;
      char* l = Lb + (isB << 15) + (s << 12) + off;
#if HAVE_ASYNC
      __builtin_amdgcn_global_load_async_to_lds_b128(
          (AS1 v4i*)g, (AS3 v4i*)l, 0, 0);
#else
      *(v16bf*)l = *(const v16bf*)g;        // fallback: load + ds_store
#endif
    }
    __builtin_prefetch((const void*)(Ag + 8192), 0, 3);
    __builtin_prefetch((const void*)(Bg + 8192), 0, 3);
  };

  stage(0, 0);
  for (int ch = 0; ch < 16; ++ch) {
#if HAVE_ASYNC
#if __has_builtin(__builtin_amdgcn_s_wait_asynccnt)
    __builtin_amdgcn_s_wait_asynccnt(0);
#else
    asm volatile("s_wait_asynccnt 0x0" ::: "memory");
#endif
#endif
    __syncthreads();                        // everyone's copies visible
    if (ch < 15) stage(ch + 1, (ch + 1) & 1);

    const char* Ab = lds + (ch & 1) * 49152;
    const char* Bb = Ab + 32768;
#pragma unroll
    for (int kc = 0; kc < 4; ++kc) {
      v16bf a0 = *(const v16bf*)(Ab + ((mw * 2 + 0) << 12) + (kc << 10) + lane * 32);
      v16bf a1 = *(const v16bf*)(Ab + ((mw * 2 + 1) << 12) + (kc << 10) + lane * 32);
      v16bf b0 = *(const v16bf*)(Bb + ((nw * 2 + 0) << 12) + (kc << 10) + lane * 32);
      v16bf b1 = *(const v16bf*)(Bb + ((nw * 2 + 1) << 12) + (kc << 10) + lane * 32);
      c00 = __builtin_amdgcn_wmma_f32_16x16x32_bf16(false, a0, false, b0, (short)0, c00, false, false);
      c01 = __builtin_amdgcn_wmma_f32_16x16x32_bf16(false, a0, false, b1, (short)0, c01, false, false);
      c10 = __builtin_amdgcn_wmma_f32_16x16x32_bf16(false, a1, false, b0, (short)0, c10, false, false);
      c11 = __builtin_amdgcn_wmma_f32_16x16x32_bf16(false, a1, false, b1, (short)0, c11, false, false);
    }
  }

  // C/D layout: lane n = lane&15; VGPR v holds M = v + 8*(lane>>4)
  if (mt0 < NMT) {                          // padded tiles (mt 124..127) skip store
    const int hi = lane >> 4, nn = lane & 15;
    v8f accs[2][2] = {{c00, c01}, {c10, c11}};
#pragma unroll
    for (int mi = 0; mi < 2; ++mi) {
#pragma unroll
      for (int ni = 0; ni < 2; ++ni) {
        v8f a = accs[mi][ni];
        int l = (nt0 + ni) * 16 + nn;
#pragma unroll
        for (int v = 0; v < 8; ++v) {
          int row = (mt0 + mi) * 16 + v + hi * 8;   // = o*62 + vnode
          int o = row / NV;
          out[((size_t)nz * NMROW + row) * NL + l] = a[v] + bias[o];
        }
      }
    }
  }
}

// ---------------------------------------------------------------------------
extern "C" void kernel_launch(void* const* d_in, const int* in_sizes, int n_in,
                              void* d_out, int out_size, void* d_ws, size_t ws_size,
                              hipStream_t stream) {
  (void)in_sizes; (void)n_in; (void)out_size; (void)ws_size;
  const float* x       = (const float*)d_in[0];
  const float* adj_PLI = (const float*)d_in[1];
  const float* adj_buf = (const float*)d_in[2];
  const float* gate_w1 = (const float*)d_in[3];
  const float* gate_w2 = (const float*)d_in[4];
  const float* mlp_w   = (const float*)d_in[5];
  const float* mlp_b   = (const float*)d_in[6];
  float* out = (float*)d_out;

  char* ws = (char*)d_ws;
  float* adj_ds      = (float*)(ws + 0);            // 3844 f32
  float* Mm          = (float*)(ws + 16384);        // 3844 f32
  float* Pg          = (float*)(ws + 32768);        // 4 * 4096 f32 (M^1..M^4)
  unsigned short* Tf = (unsigned short*)(ws + 98304);    // 128*64*512 bf16 = 8 MB
  unsigned short* Bf = (unsigned short*)(ws + 8486912);  // 32*32*64*512 bf16 = 64 MB

  gate_kernel<<<1, 256, 0, stream>>>(adj_buf, gate_w1, gate_w2, adj_ds);
  adjm_kernel<<<1, 256, 0, stream>>>(adj_PLI, adj_ds, Mm);
  pow_kernel<<<1, 256, 0, stream>>>(Mm, Pg);
  build_t<<<16384, 256, 0, stream>>>(Pg, mlp_w, Tf);      // 128*64*512 / 256
  swizzle_b<<<131072, 256, 0, stream>>>(x, Bf);           // 32*32*64*512 / 256

  dim3 grid(16, 8, 32);   // 128/8 mt-groups, 8 nt-groups, 32 batches
  gemm_wmma<<<grid, 256, 98304, stream>>>(Tf, Bf, mlp_b, out);
}
